// SelfAttention_14534169330332
// MI455X (gfx1250) — compile-verified, matrix operand order checked by reading, not run on hardware
//
#include <hip/hip_runtime.h>
#include <hip/hip_bf16.h>

typedef __bf16 bf16_t;
typedef __attribute__((ext_vector_type(16))) __bf16 v16bf;
typedef __attribute__((ext_vector_type(8)))  __bf16 v8bf;
typedef __attribute__((ext_vector_type(8)))  float   v8f;
typedef __attribute__((ext_vector_type(4)))  unsigned int v4u;
typedef __attribute__((ext_vector_type(4)))  int v4i;

#define N_EMBD 1024
#define N_HEAD 16
#define HEADD  64
#define SEQT   2048
#define BATCH  2

// ---------------------------------------------------------------------------
// CDNA5 async global->LDS staging (ASYNCcnt-tracked), with safe fallback.
// Builtin signature (from probe diagnostic): first arg = AS1 v4i*, non-const.
// ---------------------------------------------------------------------------
#if defined(__has_builtin)
#if __has_builtin(__builtin_amdgcn_global_load_async_to_lds_b128)
#define USE_ASYNC_LDS 1
#endif
#endif

typedef __attribute__((address_space(1))) v4i* gv4i_p;
typedef __attribute__((address_space(3))) v4i* lv4i_p;

__device__ __forceinline__ void copy16_async(const void* g, void* l) {
#if defined(USE_ASYNC_LDS)
    __builtin_amdgcn_global_load_async_to_lds_b128(
        (gv4i_p)g, (lv4i_p)l, 0, 0);
#else
    *(v4u*)l = *(const v4u*)g;
#endif
}

__device__ __forceinline__ void wait_async_copies() {
#if defined(USE_ASYNC_LDS)
#if __has_builtin(__builtin_amdgcn_s_wait_asynccnt)
    __builtin_amdgcn_s_wait_asynccnt(0);
#else
    asm volatile("s_wait_asynccnt 0" ::: "memory");
#endif
#endif
}

// ---------------------------------------------------------------------------
// fragment helpers
// ---------------------------------------------------------------------------
__device__ __forceinline__ bf16_t f2bf(float f) { return (bf16_t)f; }

// 16x32 bf16 A/B fragment for v_wmma_f32_16x16x32_bf16.
// ISA 7.12.2: lane holds two contiguous 8-element (16B) chunks at
// K = (lane/16)*8 and K+16.
__device__ __forceinline__ v16bf make_frag(const bf16_t* p0, const bf16_t* p1) {
    v8bf lo = *(const v8bf*)p0;
    v8bf hi = *(const v8bf*)p1;
    v16bf f;
#pragma unroll
    for (int i = 0; i < 8; ++i) { f[i] = lo[i]; f[8 + i] = hi[i]; }
    return f;
}

__device__ __forceinline__ v16bf frag_ld(const bf16_t* p0, const bf16_t* p1) {
    return make_frag(p0, p1);
}
__device__ __forceinline__ v16bf frag_ld(const float* p0, const float* p1) {
    v8f a = *(const v8f*)p0;
    v8f b = *(const v8f*)p1;
    v16bf f;
#pragma unroll
    for (int i = 0; i < 8; ++i) { f[i] = f2bf(a[i]); f[8 + i] = f2bf(b[i]); }
    return f;
}

__device__ __forceinline__ v8f wmma_bf16(v16bf a, v16bf b, v8f c) {
    // (neg_a, A, neg_b, B, c_mod, C, reuse_a, reuse_b)
    return __builtin_amdgcn_wmma_f32_16x16x32_bf16(false, a, false, b,
                                                   (short)0, c, false, false);
}

// ---------------------------------------------------------------------------
// GEMM:  out[M,N] = A[M,K] * W[N,K]^T + bias[N]
// Block = 128x128 tile, BK=32, 256 threads = 8 waves in 4(M) x 2(N).
// Tiles staged via async global->LDS copies in source precision; f32->bf16
// conversion happens in fragment build (co-executes with the matrix pipe).
// ---------------------------------------------------------------------------
template <typename TIN, typename TOUT>
__global__ __launch_bounds__(256) void gemm_xwt_kernel(
        const TIN* __restrict__ A, const float* __restrict__ W,
        const float* __restrict__ bias, TOUT* __restrict__ out,
        int M, int N, int K) {
    constexpr int BM = 128, BN = 128, BK = 32, PAD = 40;  // PAD in elements
    __shared__ TIN   sA[BM * PAD];   // A tile in source precision
    __shared__ float sB[BN * PAD];   // W tile (always f32)

    const int t    = threadIdx.x;
    const int lane = t & 31;
    const int wave = t >> 5;
    const int wm   = wave & 3;   // M sub-block of 32 rows
    const int wn   = wave >> 2;  // N sub-block of 64 cols
    const int l16  = lane & 15;
    const int half = lane >> 4;

    const int mbase = blockIdx.x * BM;
    const int nbase = blockIdx.y * BN;

    v8f c[2][4];
#pragma unroll
    for (int mt = 0; mt < 2; ++mt)
#pragma unroll
        for (int nt = 0; nt < 4; ++nt) c[mt][nt] = {};

    constexpr int CHA  = 16 / (int)sizeof(TIN); // elements per 16B chunk
    constexpr int CPRA = BK / CHA;              // chunks per row (A)
    constexpr int ITA  = (BM * CPRA) / 256;     // chunk iters per thread (A)

    for (int kb = 0; kb < K; kb += BK) {
        // ---- stage A tile (async, 16B granules) ----
#pragma unroll
        for (int i = 0; i < ITA; ++i) {
            int fid = t + i * 256;
            int row = fid / CPRA;
            int kc  = (fid % CPRA) * CHA;
            copy16_async(A + (size_t)(mbase + row) * K + kb + kc,
                         &sA[row * PAD + kc]);
        }
        // ---- stage B tile (async, 4 f32 per 16B chunk) ----
#pragma unroll
        for (int i = 0; i < 4; ++i) {
            int fid = t + i * 256;
            int row = fid >> 3;
            int kc  = (fid & 7) * 4;
            copy16_async(W + (size_t)(nbase + row) * K + kb + kc,
                         &sB[row * PAD + kc]);
        }
        wait_async_copies();
        __syncthreads();

        v16bf af[2], bfr[4];
#pragma unroll
        for (int mt = 0; mt < 2; ++mt) {
            const TIN* p = &sA[(wm * 32 + mt * 16 + l16) * PAD + half * 8];
            af[mt] = frag_ld(p, p + 16);
        }
#pragma unroll
        for (int nt = 0; nt < 4; ++nt) {
            const float* p = &sB[(wn * 64 + nt * 16 + l16) * PAD + half * 8];
            bfr[nt] = frag_ld(p, p + 16);
        }
#pragma unroll
        for (int mt = 0; mt < 2; ++mt)
#pragma unroll
            for (int nt = 0; nt < 4; ++nt)
                c[mt][nt] = wmma_bf16(af[mt], bfr[nt], c[mt][nt]);
        __syncthreads();
    }

    // ---- epilogue: bias add + store.  C/D layout: VGPR r -> M = r + 8*half,
    // N = lane%16 ----
#pragma unroll
    for (int nt = 0; nt < 4; ++nt) {
        int n = nbase + wn * 64 + nt * 16 + l16;
        float bv = bias[n];
#pragma unroll
        for (int mt = 0; mt < 2; ++mt) {
            int rbase = mbase + wm * 32 + mt * 16 + half * 8;
#pragma unroll
            for (int r = 0; r < 8; ++r) {
                float v = c[mt][nt][r] + bv;
                out[(size_t)(rbase + r) * N + n] = (TOUT)v;
            }
        }
    }
}

// ---------------------------------------------------------------------------
// Flash attention (causal).  Q/K/V: bf16 [B*T, C], head slice h*D..+D.
// 8 waves/block, wave = 16 query rows; grid = (B*H, T/128).
// Key blocks of 32 (= WMMA K).  K tile async-staged to LDS (shared by all 8
// waves); V^T staged transposed; P shuffled through LDS into A-frag layout.
// ---------------------------------------------------------------------------
__global__ __launch_bounds__(256) void attn_kernel(
        const bf16_t* __restrict__ Qb, const bf16_t* __restrict__ Kb,
        const bf16_t* __restrict__ Vb, bf16_t* __restrict__ Yb) {
    constexpr int T = SEQT, C = N_EMBD, D = HEADD, H = N_HEAD;
    constexpr int VP = 40;   // pad stride for sVt / sP
    constexpr int KP = 72;   // pad stride for sK (64 d + 8)
    const int bh   = blockIdx.x;
    const int b    = bh / H;
    const int h    = bh % H;
    const int qblk = blockIdx.y;
    const int t    = threadIdx.x;
    const int lane = t & 31;
    const int wave = t >> 5;
    const int l16  = lane & 15;
    const int half = lane >> 4;
    const int qt   = qblk * 128 + wave * 16;

    __shared__ bf16_t sK[32 * KP];          // K tile: [key][d]
    __shared__ bf16_t sVt[D * VP];          // V^T  : [d][key]
    __shared__ bf16_t sP[8][16 * VP];       // per-wave P tile 16x32

    // Q fragments (16 rows x 64 d -> two K=32 fragments), direct from global
    const size_t qrow = (size_t)(b * T + qt + l16) * C + h * D;
    v16bf qf[2];
#pragma unroll
    for (int f = 0; f < 2; ++f) {
        const bf16_t* p = Qb + qrow + f * 32 + half * 8;
        qf[f] = make_frag(p, p + 16);
    }

    v8f o[4];
#pragma unroll
    for (int nt = 0; nt < 4; ++nt) o[nt] = {};
    float mrow[8], lrow[8];
#pragma unroll
    for (int r = 0; r < 8; ++r) { mrow[r] = -1e30f; lrow[r] = 0.0f; }

    const int   qmax    = qt + 15;
    const int   nblocks = (qblk + 1) * 4;
    const float scale   = 0.125f;           // 1/sqrt(64)

    for (int j = 0; j < nblocks; ++j) {
        const int kb = j * 32;
        __syncthreads();
        {   // async-stage K tile: 32 keys x 64 d = 256 x 16B chunks
            int row = t >> 3;          // 0..31
            int dc  = (t & 7) * 8;     // 0..56
            copy16_async(Kb + (size_t)(b * T + kb + row) * C + h * D + dc,
                         &sK[row * KP + dc]);
        }
        {   // stage V^T (64 d x 32 keys), transposed scalar writes
            int d  = t & 63;
            int k0 = (t >> 6) * 8;
            const bf16_t* vsrc = Vb + (size_t)(b * T + kb + k0) * C + h * D + d;
            bf16_t* vdst = &sVt[d * VP + k0];
#pragma unroll
            for (int i = 0; i < 8; ++i) vdst[i] = vsrc[(size_t)i * C];
        }
        wait_async_copies();
        __syncthreads();
        if (kb > qmax) continue;            // wave-uniform; no barriers below

        // ---- scores: S = Q * K^T over D=64 (two chained WMMAs each) ----
        v8f s0 = {}, s1 = {};
#pragma unroll
        for (int f = 0; f < 2; ++f) {
            const bf16_t* p = &sK[l16 * KP + f * 32 + half * 8];
            s0 = wmma_bf16(qf[f], make_frag(p, p + 16), s0);
        }
#pragma unroll
        for (int f = 0; f < 2; ++f) {
            const bf16_t* p = &sK[(16 + l16) * KP + f * 32 + half * 8];
            s1 = wmma_bf16(qf[f], make_frag(p, p + 16), s1);
        }

        // ---- online softmax (row = r + 8*half, lanes l16 = N) ----
        float p0[8], p1[8];
#pragma unroll
        for (int r = 0; r < 8; ++r) {
            int qrowi = qt + r + half * 8;
            float a  = s0[r] * scale;
            float b2 = s1[r] * scale;
            if (kb + l16 > qrowi)      a  = -1e30f;
            if (kb + 16 + l16 > qrowi) b2 = -1e30f;
            float mx = fmaxf(a, b2);
#pragma unroll
            for (int off = 1; off < 16; off <<= 1)
                mx = fmaxf(mx, __shfl_xor(mx, off, 32));
            float mnew = fmaxf(mrow[r], mx);
            float ea = __expf(a - mnew);
            float eb = __expf(b2 - mnew);
            float rs = ea + eb;
#pragma unroll
            for (int off = 1; off < 16; off <<= 1)
                rs += __shfl_xor(rs, off, 32);
            float corr = __expf(mrow[r] - mnew);
            lrow[r] = lrow[r] * corr + rs;
            mrow[r] = mnew;
            p0[r] = ea; p1[r] = eb;
#pragma unroll
            for (int nt = 0; nt < 4; ++nt) o[nt][r] *= corr;
        }

        // ---- write P (C/D layout) to LDS, re-read as A fragment ----
        bf16_t* pw = &sP[wave][0];
#pragma unroll
        for (int r = 0; r < 8; ++r) {
            pw[(r + half * 8) * VP + l16]      = f2bf(p0[r]);
            pw[(r + half * 8) * VP + 16 + l16] = f2bf(p1[r]);
        }
        const bf16_t* pr = &sP[wave][l16 * VP + half * 8];
        v16bf pf = make_frag(pr, pr + 16);

        // ---- O += P * V  (B fragments from transposed V in LDS) ----
#pragma unroll
        for (int nt = 0; nt < 4; ++nt) {
            const bf16_t* vb = &sVt[(nt * 16 + l16) * VP + half * 8];
            o[nt] = wmma_bf16(pf, make_frag(vb, vb + 16), o[nt]);
        }
    }

    // ---- normalize and store ----
    const size_t orow = (size_t)(b * T + qt) * C + h * D;
#pragma unroll
    for (int nt = 0; nt < 4; ++nt)
#pragma unroll
        for (int r = 0; r < 8; ++r) {
            float v = o[nt][r] / lrow[r];
            Yb[orow + (size_t)(r + half * 8) * C + nt * 16 + l16] = f2bf(v);
        }
}

// ---------------------------------------------------------------------------
// launcher
// ---------------------------------------------------------------------------
extern "C" void kernel_launch(void* const* d_in, const int* in_sizes, int n_in,
                              void* d_out, int out_size, void* d_ws, size_t ws_size,
                              hipStream_t stream) {
    (void)in_sizes; (void)n_in; (void)out_size; (void)ws_size;
    const float* x  = (const float*)d_in[0];
    const float* Wq = (const float*)d_in[1];
    const float* bq = (const float*)d_in[2];
    const float* Wk = (const float*)d_in[3];
    const float* bk = (const float*)d_in[4];
    const float* Wv = (const float*)d_in[5];
    const float* bv = (const float*)d_in[6];
    const float* Wo = (const float*)d_in[7];
    const float* bo = (const float*)d_in[8];
    float* out = (float*)d_out;

    const int M = BATCH * SEQT;   // 4096
    const int C = N_EMBD;         // 1024
    const size_t PLANE = (size_t)M * C;

    bf16_t* Qb = (bf16_t*)d_ws;
    bf16_t* Kb = Qb + PLANE;
    bf16_t* Vb = Kb + PLANE;
    bf16_t* Yb = Vb + PLANE;

    dim3 gg(M / 128, C / 128), gb(256);
    gemm_xwt_kernel<float, bf16_t><<<gg, gb, 0, stream>>>(x, Wq, bq, Qb, M, C, C);
    gemm_xwt_kernel<float, bf16_t><<<gg, gb, 0, stream>>>(x, Wk, bk, Kb, M, C, C);
    gemm_xwt_kernel<float, bf16_t><<<gg, gb, 0, stream>>>(x, Wv, bv, Vb, M, C, C);

    dim3 ga(BATCH * N_HEAD, SEQT / 128);
    attn_kernel<<<ga, gb, 0, stream>>>(Qb, Kb, Vb, Yb);

    gemm_xwt_kernel<bf16_t, float><<<gg, gb, 0, stream>>>(Yb, Wo, bo, out, M, C, C);
}
